// ChatDecoder_36747740184918
// MI455X (gfx1250) — matrix operand order repaired and checked by reading, not compiled, not run on hardware
//
#include <hip/hip_runtime.h>
#include <hip/hip_bf16.h>

// ---- model constants (match reference) ----
#define LAY 4
#define HEADS 16
#define DM 1024
#define HD 64
#define VOC 32000
#define BB 2
#define TT 2048
#define ROWS (BB*TT)          // 4096

typedef __attribute__((ext_vector_type(16))) __bf16 v16bf;
typedef __attribute__((ext_vector_type(8)))  __bf16 v8bf;
typedef __attribute__((ext_vector_type(8)))  float  v8f;

#define DEV static __device__ __forceinline__

#if defined(__HIP_DEVICE_COMPILE__)
DEV v16bf cat8(v8bf a, v8bf b) {
  return __builtin_shufflevector(a, b, 0,1,2,3,4,5,6,7,8,9,10,11,12,13,14,15);
}
DEV v8f wmma_bf16(v16bf a, v16bf b, v8f c) {
  return __builtin_amdgcn_wmma_f32_16x16x32_bf16(false, a, false, b, (short)0, c, false, false);
}
DEV v8f vzero8() {
  v8f z;
  #pragma unroll
  for (int i = 0; i < 8; ++i) z[i] = 0.0f;
  return z;
}
// A-fragment (16x32, MxK): lane holds row, chunks at k=hi*8 and 16+hi*8
DEV v16bf frag_a(const __bf16* rowp, int hi) {
  return cat8(*(const v8bf*)(rowp + hi*8), *(const v8bf*)(rowp + 16 + hi*8));
}
// B-fragment (32x16, KxN): lane holds column, chunks at k=hi*8 and 16+hi*8
DEV v16bf frag_b(const __bf16* rowp, int hi) {
  return cat8(*(const v8bf*)(rowp + hi*8), *(const v8bf*)(rowp + 16 + hi*8));
}
DEV float rmax16(float v) {
  #pragma unroll
  for (int m = 1; m < 16; m <<= 1) v = fmaxf(v, __shfl_xor(v, m, 32));
  return v;
}
DEV float rsum16(float v) {
  #pragma unroll
  for (int m = 1; m < 16; m <<= 1) v += __shfl_xor(v, m, 32);
  return v;
}
// CDNA5 async global->LDS copy (16B per lane), tracked by ASYNCcnt
DEV void async_ld16(unsigned lds_off, const __bf16* g) {
  asm volatile("global_load_async_to_lds_b128 %0, %1, off"
               :: "v"(lds_off), "v"(g) : "memory");
}
DEV void wait_async0() { asm volatile("s_wait_asynccnt 0x0" ::: "memory"); }
DEV unsigned lds_off32(const void* p) { return (unsigned)(unsigned long long)p; }
#endif

// ---------------- f32 -> bf16 conversion ----------------
__global__ void cvt_f32_bf16(const float* __restrict__ in, __bf16* __restrict__ out, size_t n) {
#if defined(__HIP_DEVICE_COMPILE__)
  for (size_t i = (size_t)blockIdx.x * blockDim.x + threadIdx.x; i < n;
       i += (size_t)gridDim.x * blockDim.x)
    out[i] = (__bf16)in[i];
#endif
}

// ---------------- embedding: x = tok[id] + pos[t] + ctx[b] ----------------
__global__ void embed_kernel(const int* __restrict__ ids, const float* __restrict__ ctx,
                             const float* __restrict__ tok, const float* __restrict__ pos,
                             float* __restrict__ x) {
#if defined(__HIP_DEVICE_COMPILE__)
  int row = blockIdx.x;               // 0..ROWS-1
  int b = row / TT, t = row % TT;
  int id = ids[row];
  for (int d = threadIdx.x; d < DM; d += blockDim.x)
    x[(size_t)row*DM + d] = tok[(size_t)id*DM + d] + pos[(size_t)t*DM + d] + ctx[(size_t)b*DM + d];
#endif
}

// ---------------- wave-per-row LayerNorm -> bf16 ----------------
__global__ __launch_bounds__(256) void ln_bf16_kernel(const float* __restrict__ x,
                                                      const float* __restrict__ w,
                                                      const float* __restrict__ bia,
                                                      __bf16* __restrict__ out) {
#if defined(__HIP_DEVICE_COMPILE__)
  int row  = blockIdx.x * 8 + (threadIdx.x >> 5);
  int lane = threadIdx.x & 31;
  const float* xr = x + (size_t)row * DM;
  float v[32]; float s = 0.f, sq = 0.f;
  #pragma unroll
  for (int i = 0; i < 32; ++i) { float t = xr[lane + 32*i]; v[i] = t; s += t; sq += t*t; }
  #pragma unroll
  for (int m = 1; m < 32; m <<= 1) { s += __shfl_xor(s, m, 32); sq += __shfl_xor(sq, m, 32); }
  float mean = s * (1.0f/DM);
  float var  = sq * (1.0f/DM) - mean*mean;
  float rstd = rsqrtf(var + 1e-5f);
  #pragma unroll
  for (int i = 0; i < 32; ++i) {
    int d = lane + 32*i;
    out[(size_t)row*DM + d] = (__bf16)((v[i] - mean) * rstd * w[d] + bia[d]);
  }
#endif
}

// ---------------- tiled WMMA GEMM: C[M,N] = A[M,K] @ W[N,K]^T + bias ----------------
// block tile 128(M) x 128(N) x 32(K); 8 waves, each wave 16 rows x 128 cols (8 WMMA/K-step)
// mode 0: outF = val ; mode 1: outF += val (residual) ; mode 2: outB = bf16(gelu(val))
__global__ __launch_bounds__(256) void gemm_bf16_kernel(const __bf16* __restrict__ A,
                                                        const __bf16* __restrict__ W,
                                                        const float* __restrict__ bias,
                                                        float* __restrict__ outF,
                                                        __bf16* __restrict__ outB,
                                                        int N, int K, int mode) {
#if defined(__HIP_DEVICE_COMPILE__)
  __shared__ __align__(16) __bf16 As[128*32];
  __shared__ __align__(16) __bf16 Bs[128*32];
  int tid = threadIdx.x, wv = tid >> 5, lane = tid & 31, lo = lane & 15, hi = lane >> 4;
  int m0 = blockIdx.y * 128, n0 = blockIdx.x * 128;
  v8f acc[8];
  #pragma unroll
  for (int j = 0; j < 8; ++j) acc[j] = vzero8();

  for (int kk = 0; kk < K; kk += 32) {
    // async-stage A (128x32) and B (128x32): 512 x 16B chunks each, 2+2 per thread
    #pragma unroll
    for (int i = 0; i < 2; ++i) {
      int ch = tid + i*256, r = ch >> 2, c = ch & 3;
      async_ld16(lds_off32(As + r*32 + c*8), A + (size_t)(m0 + r)*K + kk + c*8);
      async_ld16(lds_off32(Bs + r*32 + c*8), W + (size_t)(n0 + r)*K + kk + c*8);
    }
    if (kk + 32 < K) {                          // global_prefetch_b8 of next K tile
      __builtin_prefetch(A + (size_t)(m0 + (tid >> 1))*K + kk + 32, 0, 1);
      __builtin_prefetch(W + (size_t)(n0 + (tid >> 1))*K + kk + 32, 0, 1);
    }
    wait_async0();        // our lanes' copies landed in LDS
    __syncthreads();      // everyone's copies visible

    v16bf af = frag_a(As + (wv*16 + lo)*32, hi);
    #pragma unroll
    for (int j = 0; j < 8; ++j) {
      v16bf bf_ = frag_b(Bs + (j*16 + lo)*32, hi);
      acc[j] = wmma_bf16(af, bf_, acc[j]);
    }
    __syncthreads();      // done reading before next iteration overwrites
  }

  #pragma unroll
  for (int j = 0; j < 8; ++j) {
    int n = n0 + j*16 + lo;
    float bv = bias ? bias[n] : 0.0f;
    #pragma unroll
    for (int r = 0; r < 8; ++r) {
      int m = m0 + wv*16 + r + 8*hi;
      float val = acc[j][r] + bv;
      size_t oi = (size_t)m * N + n;
      if (mode == 1)      outF[oi] += val;
      else if (mode == 0) outF[oi]  = val;
      else {
        float g = 0.5f * val * (1.0f + erff(val * 0.70710678118f));
        outB[oi] = (__bf16)g;
      }
    }
  }
#endif
}

// ---------------- split qkv (f32) -> q/k bf16 [b,h,t,d], v bf16 transposed [b,h,d,t] ----------------
__global__ void split_qkv_kernel(const float* __restrict__ qkv, __bf16* __restrict__ q,
                                 __bf16* __restrict__ k, __bf16* __restrict__ vt) {
#if defined(__HIP_DEVICE_COMPILE__)
  size_t n = (size_t)ROWS * 3 * DM;
  for (size_t i = (size_t)blockIdx.x * blockDim.x + threadIdx.x; i < n;
       i += (size_t)gridDim.x * blockDim.x) {
    int row = (int)(i / (3*DM));
    int c   = (int)(i % (3*DM));
    int which = c / DM, d0 = c % DM;
    int h = d0 >> 6, d = d0 & 63;
    int b = row / TT, t = row % TT;
    __bf16 v = (__bf16)qkv[i];
    if (which == 0)      q [((size_t)(b*HEADS + h)*TT + t)*HD + d] = v;
    else if (which == 1) k [((size_t)(b*HEADS + h)*TT + t)*HD + d] = v;
    else                 vt[((size_t)(b*HEADS + h)*HD + d)*TT + t] = v;
  }
#endif
}

// ---------------- flash attention: one wave per (b,h,16-query tile) ----------------
__global__ __launch_bounds__(256) void flash_kernel(const __bf16* __restrict__ q,
                                                    const __bf16* __restrict__ k,
                                                    const __bf16* __restrict__ vt,
                                                    __bf16* __restrict__ y) {
#if defined(__HIP_DEVICE_COMPILE__)
  __shared__ __align__(16) __bf16 Ps[8][16*32];   // per-wave P staging
  int tid = threadIdx.x, wv = tid >> 5, lane = tid & 31, lo = lane & 15, hi = lane >> 4;
  int wid = blockIdx.x * 8 + wv;                  // 0..4095
  int qt = wid & 127;                              // T/16 = 128
  int h  = (wid >> 7) & 15;
  int b  = wid >> 11;
  int q0 = qt * 16;

  const __bf16* qp = q + ((size_t)(b*HEADS + h)*TT + q0 + lo)*HD;
  v16bf aQ0 = cat8(*(const v8bf*)(qp +       hi*8), *(const v8bf*)(qp + 16 + hi*8));
  v16bf aQ1 = cat8(*(const v8bf*)(qp + 32 +  hi*8), *(const v8bf*)(qp + 48 + hi*8));

  v8f o0 = vzero8(), o1 = vzero8(), o2 = vzero8(), o3 = vzero8();
  float rm[8], rl[8];
  #pragma unroll
  for (int r = 0; r < 8; ++r) { rm[r] = -3.0e38f; rl[r] = 0.0f; }

  const __bf16* kb = k  + (size_t)(b*HEADS + h)*TT*HD;
  const __bf16* vb = vt + (size_t)(b*HEADS + h)*HD*TT;
  __bf16* pbuf = &Ps[wv][0];
  const float scale = 0.125f;   // 1/sqrt(64)

  for (int k0 = 0; k0 < q0 + 16; k0 += 32) {
    // scores: S = Q @ K^T  (two 16-key subtiles)
    const __bf16* kr1 = kb + (size_t)(k0 + lo)*HD;
    const __bf16* kr2 = kb + (size_t)(k0 + 16 + lo)*HD;
    v8f s1 = vzero8(), s2 = vzero8();
    s1 = wmma_bf16(aQ0, frag_b(kr1,      hi), s1);
    s1 = wmma_bf16(aQ1, frag_b(kr1 + 32, hi), s1);
    s2 = wmma_bf16(aQ0, frag_b(kr2,      hi), s2);
    s2 = wmma_bf16(aQ1, frag_b(kr2 + 32, hi), s2);

    // causal mask + online softmax
    #pragma unroll
    for (int r = 0; r < 8; ++r) {
      int qg = q0 + r + 8*hi;
      float e1 = s1[r] * scale; if (k0 + lo      > qg) e1 = -3.0e38f;
      float e2 = s2[r] * scale; if (k0 + 16 + lo > qg) e2 = -3.0e38f;
      float mx = rmax16(fmaxf(e1, e2));
      float nm = fmaxf(rm[r], mx);
      float cf = __expf(rm[r] - nm);
      float p1 = __expf(e1 - nm), p2 = __expf(e2 - nm);
      float rs = rsum16(p1 + p2);
      rl[r] = rl[r]*cf + rs; rm[r] = nm;
      o0[r] *= cf; o1[r] *= cf; o2[r] *= cf; o3[r] *= cf;
      pbuf[(r + 8*hi)*32 + lo]      = (__bf16)p1;
      pbuf[(r + 8*hi)*32 + 16 + lo] = (__bf16)p2;
    }
    // P: C-layout -> A-layout through LDS
    v16bf aP = cat8(*(const v8bf*)(pbuf + lo*32 + hi*8),
                    *(const v8bf*)(pbuf + lo*32 + 16 + hi*8));
    // O += P @ V   (V transposed: contiguous keys per lane)
    o0 = wmma_bf16(aP, frag_b(vb + (size_t)(     lo)*TT + k0, hi), o0);
    o1 = wmma_bf16(aP, frag_b(vb + (size_t)(16 + lo)*TT + k0, hi), o1);
    o2 = wmma_bf16(aP, frag_b(vb + (size_t)(32 + lo)*TT + k0, hi), o2);
    o3 = wmma_bf16(aP, frag_b(vb + (size_t)(48 + lo)*TT + k0, hi), o3);
  }

  __bf16* yb = y + (size_t)b * TT * DM;
  #pragma unroll
  for (int r = 0; r < 8; ++r) {
    int m = q0 + r + 8*hi;
    float inv = 1.0f / rl[r];
    size_t base = (size_t)m * DM + h*64 + lo;
    yb[base]      = (__bf16)(o0[r] * inv);
    yb[base + 16] = (__bf16)(o1[r] * inv);
    yb[base + 32] = (__bf16)(o2[r] * inv);
    yb[base + 48] = (__bf16)(o3[r] * inv);
  }
#endif
}

// ================================================================
extern "C" void kernel_launch(void* const* d_in, const int* in_sizes, int n_in,
                              void* d_out, int out_size, void* d_ws, size_t ws_size,
                              hipStream_t stream) {
  (void)in_sizes; (void)n_in; (void)out_size; (void)ws_size;
  const int*   token_ids = (const int*)  d_in[0];
  const float* ctx_emb   = (const float*)d_in[1];
  const float* tok_emb   = (const float*)d_in[2];
  const float* pos_emb   = (const float*)d_in[3];
  const float* qkv_w     = (const float*)d_in[4];
  const float* qkv_b     = (const float*)d_in[5];
  const float* proj_w    = (const float*)d_in[6];
  const float* proj_b    = (const float*)d_in[7];
  const float* ln1_w     = (const float*)d_in[8];
  const float* ln1_b     = (const float*)d_in[9];
  const float* ln2_w     = (const float*)d_in[10];
  const float* ln2_b     = (const float*)d_in[11];
  const float* f1_w      = (const float*)d_in[12];
  const float* f1_b      = (const float*)d_in[13];
  const float* f2_w      = (const float*)d_in[14];
  const float* f2_b      = (const float*)d_in[15];
  const float* lnf_w     = (const float*)d_in[16];
  const float* lnf_b     = (const float*)d_in[17];
  float* logits = (float*)d_out;

  // ---- workspace layout (256B aligned blocks) ----
  size_t off = 0;
  auto take = [&](size_t bytes) -> void* {
    void* p = (char*)d_ws + off;
    off += (bytes + 255) & ~(size_t)255;
    return p;
  };
  const size_t D3 = 3*(size_t)DM, D4 = 4*(size_t)DM;
  __bf16* qkvw_bf = (__bf16*)take((size_t)LAY*D3*DM*2);
  __bf16* projw_bf= (__bf16*)take((size_t)LAY*DM*DM*2);
  __bf16* f1w_bf  = (__bf16*)take((size_t)LAY*D4*DM*2);
  __bf16* f2w_bf  = (__bf16*)take((size_t)LAY*DM*D4*2);
  __bf16* tok_bf  = (__bf16*)take((size_t)VOC*DM*2);
  float*  x       = (float*) take((size_t)ROWS*DM*4);
  __bf16* h_bf    = (__bf16*)take((size_t)ROWS*DM*2);
  float*  qkv_f32 = (float*) take((size_t)ROWS*D3*4);
  __bf16* q_bf    = (__bf16*)take((size_t)ROWS*DM*2);
  __bf16* k_bf    = (__bf16*)take((size_t)ROWS*DM*2);
  __bf16* vt_bf   = (__bf16*)take((size_t)ROWS*DM*2);
  __bf16* y_bf    = (__bf16*)take((size_t)ROWS*DM*2);
  __bf16* ff_bf   = (__bf16*)take((size_t)ROWS*D4*2);

  // ---- weight conversion to bf16 (deterministic, each call) ----
  cvt_f32_bf16<<<2048, 256, 0, stream>>>(qkv_w,  qkvw_bf, (size_t)LAY*D3*DM);
  cvt_f32_bf16<<<2048, 256, 0, stream>>>(proj_w, projw_bf,(size_t)LAY*DM*DM);
  cvt_f32_bf16<<<2048, 256, 0, stream>>>(f1_w,   f1w_bf,  (size_t)LAY*D4*DM);
  cvt_f32_bf16<<<2048, 256, 0, stream>>>(f2_w,   f2w_bf,  (size_t)LAY*DM*D4);
  cvt_f32_bf16<<<4096, 256, 0, stream>>>(tok_emb,tok_bf,  (size_t)VOC*DM);

  // ---- embedding ----
  embed_kernel<<<ROWS, 256, 0, stream>>>(token_ids, ctx_emb, tok_emb, pos_emb, x);

  dim3 blk(256);
  for (int l = 0; l < LAY; ++l) {
    // LN1 -> h_bf
    ln_bf16_kernel<<<ROWS/8, blk, 0, stream>>>(x, ln1_w + (size_t)l*DM, ln1_b + (size_t)l*DM, h_bf);
    // QKV GEMM: [ROWS,3D] fp32
    gemm_bf16_kernel<<<dim3((3*DM)/128, ROWS/128), blk, 0, stream>>>(
        h_bf, qkvw_bf + (size_t)l*D3*DM, qkv_b + (size_t)l*D3, qkv_f32, nullptr,
        3*DM, DM, 0);
    // split
    split_qkv_kernel<<<4096, blk, 0, stream>>>(qkv_f32, q_bf, k_bf, vt_bf);
    // attention
    flash_kernel<<<(BB*HEADS*(TT/16))/8, blk, 0, stream>>>(q_bf, k_bf, vt_bf, y_bf);
    // proj GEMM + residual into x
    gemm_bf16_kernel<<<dim3(DM/128, ROWS/128), blk, 0, stream>>>(
        y_bf, projw_bf + (size_t)l*DM*DM, proj_b + (size_t)l*DM, x, nullptr,
        DM, DM, 1);
    // LN2 -> h_bf
    ln_bf16_kernel<<<ROWS/8, blk, 0, stream>>>(x, ln2_w + (size_t)l*DM, ln2_b + (size_t)l*DM, h_bf);
    // FFN1 GEMM + GELU -> ff_bf
    gemm_bf16_kernel<<<dim3((4*DM)/128, ROWS/128), blk, 0, stream>>>(
        h_bf, f1w_bf + (size_t)l*D4*DM, f1_b + (size_t)l*D4, nullptr, ff_bf,
        4*DM, DM, 2);
    // FFN2 GEMM + residual into x
    gemm_bf16_kernel<<<dim3(DM/128, ROWS/128), blk, 0, stream>>>(
        ff_bf, f2w_bf + (size_t)l*DM*D4, f2_b + (size_t)l*DM, x, nullptr,
        DM, 4*DM, 1);
  }

  // final LN -> h_bf
  ln_bf16_kernel<<<ROWS/8, blk, 0, stream>>>(x, lnf_w, lnf_b, h_bf);
  // logits = h @ tok_emb^T  (no bias), fp32 out
  gemm_bf16_kernel<<<dim3(VOC/128, ROWS/128), blk, 0, stream>>>(
      h_bf, tok_bf, nullptr, logits, nullptr, VOC, DM, 0);
}
